// TuckERInteraction_20615843021592
// MI455X (gfx1250) — matrix operand order repaired and checked by reading, not compiled, not run on hardware
//
#include <hip/hip_runtime.h>

#define D_E 128
#define NROWS 32768          // B*S = 16*2048
#define QCNT 256             // NROWS / 128
#define BN_EPS 1e-5f
#define BROW 136             // padded LDS row stride (elements): 272B -> conflict-free b128

typedef __attribute__((ext_vector_type(16))) __bf16 v16bf;
typedef __attribute__((ext_vector_type(8)))  __bf16 v8bf;
typedef __attribute__((ext_vector_type(8)))  float  v8f;

__device__ __forceinline__ __bf16 f2bf(float f) {
  union { float f; unsigned u; } in; in.f = f;
  unsigned r = (in.u + 0x7FFFu + ((in.u >> 16) & 1u)) >> 16;   // RNE
  union { unsigned short s; __bf16 b; } out; out.s = (unsigned short)r;
  return out.b;
}

// CDNA5 async global->LDS copy (ASYNCcnt-tracked), GVS addressing:
//   mem = SGPR64(base) + VGPR32(byte_off);  LDS dest = per-lane VGPR lds_off.
__device__ __forceinline__ void async_ld_b128(unsigned lds_off, const void* base,
                                              int byte_off) {
  asm volatile("global_load_async_to_lds_b128 %0, %1, %2"
               :: "v"(lds_off), "v"(byte_off), "s"(base) : "memory");
}
__device__ __forceinline__ void wait_async0() {
  asm volatile("s_wait_asynccnt 0x0" ::: "memory");
}
// Low 32 bits of a flat LDS pointer == LDS byte address (ISA 10.2 aperture rule).
__device__ __forceinline__ unsigned lds_addr32(const void* p) {
  return (unsigned)(size_t)p;
}

// ---------------------------------------------------------------------------
// Prep: BatchNorm(h) -> bf16 in permuted layout Hperm[(q*128+kk)*128 + c],
//       BatchNorm(t) -> f32 natural layout.
// ---------------------------------------------------------------------------
__global__ __launch_bounds__(256) void prep_ht(
    const float* __restrict__ h, const float* __restrict__ t,
    const float* __restrict__ g0, const float* __restrict__ b0,
    const float* __restrict__ m0, const float* __restrict__ v0,
    const float* __restrict__ g1, const float* __restrict__ b1,
    const float* __restrict__ m1, const float* __restrict__ v1,
    __bf16* __restrict__ Hperm, float* __restrict__ Tbn) {
  int idx = blockIdx.x * blockDim.x + threadIdx.x;     // over NROWS*128
  int n = idx >> 7, c = idx & 127;
  float hb = (h[idx] - m0[c]) * rsqrtf(v0[c] + BN_EPS) * g0[c] + b0[c];
  int nperm = ((n & (QCNT - 1)) << 7) | (n >> 8);      // q*128 + kk
  Hperm[((size_t)nperm << 7) + c] = f2bf(hb);
  Tbn[idx] = (t[idx] - m1[c]) * rsqrtf(v1[c] + BN_EPS) * g1[c] + b1[c];
}

// ---------------------------------------------------------------------------
// LDS-tiled transpose r (NROWS x 128) -> RT (128 x NROWS), bf16.
// ---------------------------------------------------------------------------
__global__ __launch_bounds__(256) void transpose_r(const float* __restrict__ r,
                                                   __bf16* __restrict__ RT) {
  __shared__ __bf16 tile[32][33];
  int tx = threadIdx.x, ty = threadIdx.y;              // 32 x 8
  int n0 = blockIdx.x * 32, c0 = blockIdx.y * 32;
  #pragma unroll
  for (int s = 0; s < 32; s += 8)
    tile[ty + s][tx] = f2bf(r[(size_t)(n0 + ty + s) * D_E + (c0 + tx)]);
  __syncthreads();
  #pragma unroll
  for (int s = 0; s < 32; s += 8)
    RT[(size_t)(c0 + ty + s) * NROWS + (n0 + tx)] = tile[tx][ty + s];
}

// ---------------------------------------------------------------------------
// LDS-tiled per-k transpose W[k][i][j] -> WT[k][j][i], bf16.
// ---------------------------------------------------------------------------
__global__ __launch_bounds__(256) void transpose_w(const float* __restrict__ W,
                                                   __bf16* __restrict__ WT) {
  __shared__ __bf16 tile[32][33];
  int k = blockIdx.z;
  const float* Wk = W + (size_t)k * D_E * D_E;
  __bf16* WTk = WT + (size_t)k * D_E * D_E;
  int tx = threadIdx.x, ty = threadIdx.y;
  int i0 = blockIdx.x * 32, j0 = blockIdx.y * 32;
  #pragma unroll
  for (int s = 0; s < 32; s += 8)
    tile[ty + s][tx] = f2bf(Wk[(size_t)(i0 + ty + s) * D_E + (j0 + tx)]);
  __syncthreads();
  #pragma unroll
  for (int s = 0; s < 32; s += 8)
    WTk[(size_t)(j0 + ty + s) * D_E + (i0 + tx)] = tile[tx][ty + s];
}

// ---------------------------------------------------------------------------
// Stage 1: per-q 128x128x128 GEMM  U_q = H_q @ R_q  (bf16 in, f32 acc).
// B (= R_q^T rows) is staged once per WG into LDS via async global->LDS;
// 8 waves; wave w owns row-tile w (16 rows) x all 8 col-tiles.
// U stored q-major blocked: Uq[q*16384 + row*128 + col].
// ---------------------------------------------------------------------------
__global__ __launch_bounds__(256) void stage1_gemm(
    const __bf16* __restrict__ Hperm, const __bf16* __restrict__ RT,
    __bf16* __restrict__ Uq) {
  __shared__ alignas(64) __bf16 Blds[128 * BROW];   // 34 KB, row j = B^T col j
  int q = blockIdx.x;
  int tid = threadIdx.x;
  int wave = tid >> 5, lane = tid & 31;
  int half = lane >> 4;                 // 0: lanes 0-15, 1: lanes 16-31
  int lm = lane & 15;
  int kbA = half * 8;                   // A sub-K start (ISA 16-bit A layout)
  int kbB = half * 8;                   // B sub-K start (same table layout as A)

  // ---- async-stage B: 128 rows x 256B; 8 rounds x (8 waves x 2 rows) ----
  #pragma unroll
  for (int rnd = 0; rnd < 8; ++rnd) {
    int row = rnd * 16 + wave * 2 + half;            // 0..127 (col of B)
    int d = lm * 8;                                  // element within row
    unsigned loff = lds_addr32(&Blds[row * BROW + d]);
    int goff = (int)(((unsigned)row * NROWS + q * 128 + d) * sizeof(__bf16));
    async_ld_b128(loff, RT, goff);
  }
  wait_async0();          // own slice complete
  __syncthreads();        // all waves' slices visible

  const __bf16* Abase = Hperm + (((size_t)q * 128 + wave * 16 + lm) << 7);
  v8f acc[8] = {};
  #pragma unroll
  for (int k0 = 0; k0 < 128; k0 += 32) {
    v8bf alo = *(const v8bf*)(Abase + k0 + kbA);
    v8bf ahi = *(const v8bf*)(Abase + k0 + kbA + 16);
    v16bf a = __builtin_shufflevector(alo, ahi,
        0,1,2,3,4,5,6,7,8,9,10,11,12,13,14,15);
    #pragma unroll
    for (int ct = 0; ct < 8; ++ct) {
      int col = ct * 16 + lm;
      const __bf16* bp = &Blds[col * BROW + k0 + kbB];
      v8bf blo = *(const v8bf*)bp;
      v8bf bhi = *(const v8bf*)(bp + 16);
      v16bf b = __builtin_shufflevector(blo, bhi,
          0,1,2,3,4,5,6,7,8,9,10,11,12,13,14,15);
      acc[ct] = __builtin_amdgcn_wmma_f32_16x16x32_bf16(
          false, a, false, b, (short)0, acc[ct], false, false);
    }
  }

  __bf16* Ub = Uq + ((size_t)q << 14);
  int rbase = wave * 16 + half * 8;     // C/D layout: VGPR v -> row rbase+v
  #pragma unroll
  for (int ct = 0; ct < 8; ++ct)
    #pragma unroll
    for (int v = 0; v < 8; ++v)
      Ub[(rbase + v) * 128 + ct * 16 + lm] = f2bf(acc[ct][v]);
}

// ---------------------------------------------------------------------------
// Stage 2: per-k 256x128x128 GEMM  HR = U_k @ W[k], fused with the row-wise
// dot against t_bn and a 16-lane shfl_xor reduction -> scores.
// W^T[k] staged once per WG into LDS via async global->LDS.
// 8 waves; wave owns 2 row-tiles (32 rows) x 8 col-tiles.
// ---------------------------------------------------------------------------
__global__ __launch_bounds__(256) void stage2_gemm_dot(
    const __bf16* __restrict__ Uq, const __bf16* __restrict__ WT,
    const float* __restrict__ Tbn, float* __restrict__ out) {
  __shared__ alignas(64) __bf16 Blds[128 * BROW];   // 34 KB, row j = WT[k][j][*]
  int k = blockIdx.x;
  int tid = threadIdx.x;
  int wave = tid >> 5, lane = tid & 31;
  int half = lane >> 4;
  int lm = lane & 15;
  int kbA = half * 8;
  int kbB = half * 8;
  const __bf16* WTk = WT + ((size_t)k << 14);

  // ---- async-stage B = WT[k] (32 KB contiguous) into padded LDS ----
  #pragma unroll
  for (int rnd = 0; rnd < 8; ++rnd) {
    int row = rnd * 16 + wave * 2 + half;            // 0..127 (col j)
    int d = lm * 8;
    unsigned loff = lds_addr32(&Blds[row * BROW + d]);
    int goff = (int)(((unsigned)row * 128 + d) * sizeof(__bf16));
    async_ld_b128(loff, WTk, goff);
  }
  wait_async0();
  __syncthreads();

  for (int rt = 0; rt < 2; ++rt) {
    int q = wave * 32 + rt * 16 + lm;   // A row index (0..255)
    const __bf16* Abase = Uq + (((size_t)q * 128 + k) << 7);

    v8f acc[8] = {};
    #pragma unroll
    for (int k0 = 0; k0 < 128; k0 += 32) {
      v8bf alo = *(const v8bf*)(Abase + k0 + kbA);
      v8bf ahi = *(const v8bf*)(Abase + k0 + kbA + 16);
      v16bf a = __builtin_shufflevector(alo, ahi,
          0,1,2,3,4,5,6,7,8,9,10,11,12,13,14,15);
      #pragma unroll
      for (int ct = 0; ct < 8; ++ct) {
        int col = ct * 16 + lm;
        const __bf16* bp = &Blds[col * BROW + k0 + kbB];
        v8bf blo = *(const v8bf*)bp;
        v8bf bhi = *(const v8bf*)(bp + 16);
        v16bf b = __builtin_shufflevector(blo, bhi,
            0,1,2,3,4,5,6,7,8,9,10,11,12,13,14,15);
        acc[ct] = __builtin_amdgcn_wmma_f32_16x16x32_bf16(
            false, a, false, b, (short)0, acc[ct], false, false);
      }
    }

    // Epilogue: s[v] = sum_j hr[row, j] * t_bn[row, j] (partial per lane)
    int rbase = wave * 32 + rt * 16 + half * 8;
    float s[8] = {};
    #pragma unroll
    for (int ct = 0; ct < 8; ++ct) {
      int col = ct * 16 + lm;
      #pragma unroll
      for (int v = 0; v < 8; ++v) {
        int np = k * 256 + rbase + v;
        s[v] += acc[ct][v] * Tbn[((size_t)np << 7) + col];
      }
    }
    #pragma unroll
    for (int v = 0; v < 8; ++v) {
      float x = s[v];
      x += __shfl_xor(x, 1, 32);
      x += __shfl_xor(x, 2, 32);
      x += __shfl_xor(x, 4, 32);
      x += __shfl_xor(x, 8, 32);
      if (lm == 0) out[k * 256 + rbase + v] = x;
    }
  }
}

// ---------------------------------------------------------------------------
extern "C" void kernel_launch(void* const* d_in, const int* in_sizes, int n_in,
                              void* d_out, int out_size, void* d_ws, size_t ws_size,
                              hipStream_t stream) {
  const float* h  = (const float*)d_in[0];
  const float* r  = (const float*)d_in[1];
  const float* t  = (const float*)d_in[2];
  const float* W  = (const float*)d_in[3];
  const float* g0 = (const float*)d_in[4];
  const float* b0 = (const float*)d_in[5];
  const float* m0 = (const float*)d_in[6];
  const float* v0 = (const float*)d_in[7];
  const float* g1 = (const float*)d_in[8];
  const float* b1 = (const float*)d_in[9];
  const float* m1 = (const float*)d_in[10];
  const float* v1 = (const float*)d_in[11];
  float* out = (float*)d_out;

  // Workspace carve-up (44 MB total):
  //   Hperm bf16 N*128 (8MB) | RT bf16 128*N (8MB) | WT bf16 128^3 (4MB)
  //   Tbn f32 N*128 (16MB)   | Uq bf16 N*128 (8MB)
  __bf16* Hperm = (__bf16*)d_ws;
  __bf16* RT    = Hperm + (size_t)NROWS * D_E;
  __bf16* WT    = RT + (size_t)D_E * NROWS;
  float*  Tbn   = (float*)(WT + (size_t)D_E * D_E * D_E);
  __bf16* Uq    = (__bf16*)(Tbn + (size_t)NROWS * D_E);

  prep_ht<<<(NROWS * D_E) / 256, 256, 0, stream>>>(
      h, t, g0, b0, m0, v0, g1, b1, m1, v1, Hperm, Tbn);
  transpose_r<<<dim3(NROWS / 32, D_E / 32), dim3(32, 8), 0, stream>>>(r, RT);
  transpose_w<<<dim3(4, 4, D_E), dim3(32, 8), 0, stream>>>(W, WT);
  stage1_gemm<<<QCNT, 256, 0, stream>>>(Hperm, RT, Uq);
  stage2_gemm_dot<<<D_E, 256, 0, stream>>>(Uq, WT, Tbn, out);
}
